// Flow_22411139350844
// MI455X (gfx1250) — matrix operand order repaired and checked by reading, not compiled, not run on hardware
//
#include <hip/hip_runtime.h>
#include <hip/hip_bf16.h>

// ---------------------------------------------------------------------------
// Types for CDNA5 WMMA
// ---------------------------------------------------------------------------
typedef __attribute__((ext_vector_type(16))) __bf16 v16bf;
typedef __attribute__((ext_vector_type(8)))  float  v8f;

// float -> bf16 round-to-nearest-even (bit pattern)
__device__ __forceinline__ unsigned short f2bf(float f) {
    unsigned int u = __float_as_uint(f);
    unsigned int r = u + 0x7FFFu + ((u >> 16) & 1u);
    return (unsigned short)(r >> 16);
}

__device__ __forceinline__ float sigmoid_fast(float x) {
    return 1.0f / (1.0f + __expf(-x));
}
__device__ __forceinline__ float tanh_fast(float x) {
    x = fminf(fmaxf(x, -15.0f), 15.0f);
    float e = __expf(2.0f * x);
    return (e - 1.0f) / (e + 1.0f);
}

// A-fragment (16x32 bf16, row m = lane%16): ISA layout = two contiguous
// 8-half (16B) chunks at K = kt*32 + khalf*8 and K = kt*32 + 16 + khalf*8.
__device__ __forceinline__ v16bf load_afrag(const unsigned short* rowptr, int kbase) {
    union { uint4 q[2]; v16bf v; } u;
    u.q[0] = *(const uint4*)(rowptr + kbase);
    u.q[1] = *(const uint4*)(rowptr + kbase + 16);
    return u.v;
}

// B-fragment pre-packed: frag index -> 32 lanes x 16 contiguous halfs (32B/lane)
__device__ __forceinline__ v16bf load_bfrag(const unsigned short* base, size_t voff,
                                            int frag, int lane) {
    return *(const v16bf*)(base + voff + ((size_t)frag * 32 + (size_t)lane) * 16);
}

__device__ __forceinline__ v8f wmma_bf16(v16bf a, v16bf b, v8f c) {
    return __builtin_amdgcn_wmma_f32_16x16x32_bf16(false, a, false, b, (short)0, c, false, false);
}

// ---------------------------------------------------------------------------
// Problem constants
// ---------------------------------------------------------------------------
#define BATCH 4096
#define DSTEPS 64
#define HGRU 256
#define G3 768          // 3*H
#define INW 209         // 3*D + NT + 1
#define CBM 208         // 3*D + NT
#define W1COLS 464      // H + 3*D + NT

// Pack element counts
#define WHH_PACK_N (48 * 8 * 512)   // 196608
#define W1H_PACK_N (16 * 8 * 512)   // 65536
#define W2_PACK_N  (8 * 512)        // 4096
#define WIHZ_N     768

// ---------------------------------------------------------------------------
// Kernel 1: pack weights into bf16 WMMA B-fragment layout
// B-matrix (32x16 bf16): lane = nl + 16*khalf holds K = khalf*8 + (j&7) + 16*(j>>3) (j=0..15)
// contiguous halfs; element = W[n_global][k]  (B column n = W row n).
// ---------------------------------------------------------------------------
__global__ void __launch_bounds__(256) flow_pack(
    const float* __restrict__ whh, const float* __restrict__ w1,
    const float* __restrict__ w2, const float* __restrict__ wih,
    unsigned short* __restrict__ whh_p, unsigned short* __restrict__ w1h_p,
    unsigned short* __restrict__ w2_p, float* __restrict__ wihz)
{
    int i = blockIdx.x * 256 + threadIdx.x;
    if (i < WHH_PACK_N) {
        int fi = i >> 9, r = i & 511;
        int lane = r >> 4, j = r & 15;
        int nt = fi >> 3, kt = fi & 7;
        int n = nt * 16 + (lane & 15);
        int k = kt * 32 + (lane >> 4) * 8 + (j & 7) + (j >> 3) * 16;
        whh_p[i] = f2bf(whh[(size_t)n * HGRU + k]);
    } else if (i < WHH_PACK_N + W1H_PACK_N) {
        int ii = i - WHH_PACK_N;
        int fi = ii >> 9, r = ii & 511;
        int lane = r >> 4, j = r & 15;
        int nt = fi >> 3, kt = fi & 7;
        int n = nt * 16 + (lane & 15);
        int k = kt * 32 + (lane >> 4) * 8 + (j & 7) + (j >> 3) * 16;
        w1h_p[ii] = f2bf(w1[(size_t)n * W1COLS + k]);
    } else if (i < WHH_PACK_N + W1H_PACK_N + W2_PACK_N) {
        int ii = i - WHH_PACK_N - W1H_PACK_N;
        int kt = ii >> 9, r = ii & 511;
        int lane = r >> 4, j = r & 15;
        int n = lane & 15;
        int k = kt * 32 + (lane >> 4) * 8 + (j & 7) + (j >> 3) * 16;
        w2_p[ii] = (n < 9) ? f2bf(w2[(size_t)n * HGRU + k]) : (unsigned short)0;
    } else if (i < WHH_PACK_N + W1H_PACK_N + W2_PACK_N + WIHZ_N) {
        int g = i - WHH_PACK_N - W1H_PACK_N - W2_PACK_N;
        wihz[g] = wih[(size_t)g * INW];   // column 0 of gru_w_ih
    }
}

// ---------------------------------------------------------------------------
// Kernel 2: z-independent precompute per batch row:
//   constih[b][0:512]   = cbm @ W_ih[:,1:].T + b_ih + b_hh   (r,z gates)
//   constih[b][512:768] = cbm @ W_ih[:,1:].T + b_ih          (n gate)
//   cbmw1[b][:]         = cbm @ w1[:,256:].T + b1
//   cnt[b]              = sum(m*(1-b))  (mask length)
// ---------------------------------------------------------------------------
__global__ void __launch_bounds__(256) flow_prep(
    const float* __restrict__ c, const float* __restrict__ bg,
    const float* __restrict__ m, const float* __restrict__ wih,
    const float* __restrict__ bih, const float* __restrict__ bhh,
    const float* __restrict__ w1, const float* __restrict__ b1,
    float* __restrict__ constih, float* __restrict__ cbmw1,
    int* __restrict__ cnt)
{
    int row = blockIdx.x;
    int tid = threadIdx.x;
    __shared__ float cbm_s[CBM];
    for (int i = tid; i < CBM; i += 256) {
        float v;
        if (i < 80)       v = c[(size_t)row * 80 + i];
        else if (i < 144) v = bg[(size_t)row * 64 + (i - 80)];
        else              v = m[(size_t)row * 64 + (i - 144)];
        cbm_s[i] = v;
    }
    __syncthreads();
    for (int o = tid; o < G3; o += 256) {
        float s = bih[o] + ((o < 512) ? bhh[o] : 0.0f);
        const float* wr = wih + (size_t)o * INW + 1;
        #pragma unroll 8
        for (int i = 0; i < CBM; ++i) s = fmaf(cbm_s[i], wr[i], s);
        constih[(size_t)row * G3 + o] = s;
    }
    for (int o = tid; o < HGRU; o += 256) {
        float s = b1[o];
        const float* wr = w1 + (size_t)o * W1COLS + HGRU;
        #pragma unroll 8
        for (int i = 0; i < CBM; ++i) s = fmaf(cbm_s[i], wr[i], s);
        cbmw1[(size_t)row * HGRU + o] = s;
    }
    if (tid == 0) {
        int cn = 0;
        for (int t = 0; t < DSTEPS; ++t)
            cn += (m[(size_t)row * 64 + t] * (1.0f - bg[(size_t)row * 64 + t]) > 0.5f) ? 1 : 0;
        cnt[row] = cn;
    }
}

// ---------------------------------------------------------------------------
// Kernel 3: fused GRU scan + MLP + mixture log-likelihood.
// One block = 16 batch rows, 8 waves. Per step:
//   48 WMMA (GRU gh) + 16 WMMA (MLP1) + 8 WMMA (MLP2, wave 0).
// h kept in LDS as bf16; W_hh/w1/w2 B-fragments streamed from L2 each step.
// LICM on the fragment loads is defeated by passing an OFFSET (not the
// pointer) through an empty asm each kt step: loads stay per-iteration, but
// the base pointer keeps its kernarg provenance so they lower to
// global_load_b128 (LOADcnt only), not flat_load (LOADcnt+DScnt).
// ---------------------------------------------------------------------------
__global__ void __launch_bounds__(256) flow_main(
    const float* __restrict__ z, const float* __restrict__ constih,
    const float* __restrict__ cbmw1, const float* __restrict__ wihz,
    const float* __restrict__ bhh, const int* __restrict__ cnt,
    const unsigned short* __restrict__ whh_p,
    const unsigned short* __restrict__ w1h_p,
    const unsigned short* __restrict__ w2_p,
    const float* __restrict__ b2, float* __restrict__ out)
{
    __shared__ __attribute__((aligned(32))) unsigned short h_s[16][264];   // 256 + 8 pad
    __shared__ __attribute__((aligned(32))) unsigned short h1_s[16][264];
    __shared__ float z_s[16][68];
    __shared__ float params_s[16][20];

    const int tid   = threadIdx.x;
    const int lane  = tid & 31;
    const int w     = tid >> 5;        // wave id 0..7
    const int nl    = lane & 15;
    const int khalf = lane >> 4;
    const int rowOff = khalf * 8;      // C-layout: VGPR j -> row j + rowOff
    const int base  = blockIdx.x * 16;

    for (int i = tid; i < 16 * 264; i += 256) ((unsigned short*)h_s)[i] = 0;  // h0 = 0
    for (int i = tid; i < 16 * 64; i += 256)
        z_s[i >> 6][i & 63] = z[(size_t)(base + (i >> 6)) * 64 + (i & 63)];

    // Wave w owns hidden columns colA = 16w + nl and colB = 128 + 16w + nl
    const int colA = 16 * w + nl;
    const int colB = 128 + 16 * w + nl;
    const float wzr0 = wihz[colA],       wzr1 = wihz[colB];
    const float wzz0 = wihz[256 + colA], wzz1 = wihz[256 + colB];
    const float wzn0 = wihz[512 + colA], wzn1 = wihz[512 + colB];
    const float bn0  = bhh[512 + colA],  bn1  = bhh[512 + colB];
    const float pb   = (nl < 9) ? b2[nl] : 0.0f;

    // Preload per-lane loop-invariant constants into registers (64 floats):
    // const_ih columns for r/z/n gates + cbm@w1 columns.
    float cir0[8], cir1[8], ciz0[8], ciz1[8], cin0[8], cin1[8], cm0[8], cm1[8];
    {
        const float* cih = constih + (size_t)base * G3;
        const float* cw1 = cbmw1 + (size_t)base * HGRU;
        #pragma unroll
        for (int j = 0; j < 8; ++j) {
            int mr = j + rowOff;
            const float* cr = cih + (size_t)mr * G3;
            cir0[j] = cr[colA];
            cir1[j] = cr[colB];
            ciz0[j] = cr[256 + colA];
            ciz1[j] = cr[256 + colB];
            cin0[j] = cr[512 + colA];
            cin1[j] = cr[512 + colB];
            cm0[j]  = cw1[(size_t)mr * HGRU + colA];
            cm1[j]  = cw1[(size_t)mr * HGRU + colB];
        }
    }

    float hA[8], hB[8];
    #pragma unroll
    for (int j = 0; j < 8; ++j) { hA[j] = 0.0f; hB[j] = 0.0f; }

    int cn = 0; float outll = 0.0f;
    if (tid < 16) cn = cnt[base + tid];

    __syncthreads();

    #pragma unroll 1
    for (int t = 0; t < DSTEPS; ++t) {
        float zp[8];
        #pragma unroll
        for (int j = 0; j < 8; ++j)
            zp[j] = (t == 0) ? -1.0f : z_s[j + rowOff][t - 1];

        // ---- GRU: gh = h @ W_hh.T, seeded with gi (r,z) / b_hh (n) ----
        v8f a0, a1, a2, a3, a4, a5;
        #pragma unroll
        for (int j = 0; j < 8; ++j) {
            a0[j] = cir0[j] + zp[j] * wzr0;
            a1[j] = cir1[j] + zp[j] * wzr1;
            a2[j] = ciz0[j] + zp[j] * wzz0;
            a3[j] = ciz1[j] + zp[j] * wzz1;
            a4[j] = bn0;
            a5[j] = bn1;
        }
        {
            size_t voff = 0;
            #pragma unroll
            for (int kt = 0; kt < 8; ++kt) {
                asm volatile("" : "+s"(voff));   // defeat LICM, keep global AS
                v16bf A = load_afrag(&h_s[nl][0], kt * 32 + khalf * 8);
                a0 = wmma_bf16(A, load_bfrag(whh_p, voff, (w     ) * 8 + kt, lane), a0);
                a1 = wmma_bf16(A, load_bfrag(whh_p, voff, (w +  8) * 8 + kt, lane), a1);
                a2 = wmma_bf16(A, load_bfrag(whh_p, voff, (w + 16) * 8 + kt, lane), a2);
                a3 = wmma_bf16(A, load_bfrag(whh_p, voff, (w + 24) * 8 + kt, lane), a3);
                a4 = wmma_bf16(A, load_bfrag(whh_p, voff, (w + 32) * 8 + kt, lane), a4);
                a5 = wmma_bf16(A, load_bfrag(whh_p, voff, (w + 40) * 8 + kt, lane), a5);
            }
        }
        #pragma unroll
        for (int j = 0; j < 8; ++j) {
            float r0  = sigmoid_fast(a0[j]);
            float r1  = sigmoid_fast(a1[j]);
            float zg0 = sigmoid_fast(a2[j]);
            float zg1 = sigmoid_fast(a3[j]);
            float n0  = tanh_fast(cin0[j] + zp[j] * wzn0 + r0 * a4[j]);
            float n1  = tanh_fast(cin1[j] + zp[j] * wzn1 + r1 * a5[j]);
            hA[j] = (1.0f - zg0) * n0 + zg0 * hA[j];
            hB[j] = (1.0f - zg1) * n1 + zg1 * hB[j];
        }
        __syncthreads();   // all waves finished reading h_s (A-fragments)
        #pragma unroll
        for (int j = 0; j < 8; ++j) {
            int mr = j + rowOff;
            h_s[mr][colA] = f2bf(hA[j]);
            h_s[mr][colB] = f2bf(hB[j]);
        }
        __syncthreads();   // h_s now holds hs[t]

        // ---- MLP1: h1 = tanh(hs_t @ w1h.T + cbmw1) ----
        v8f m0, m1;
        #pragma unroll
        for (int j = 0; j < 8; ++j) { m0[j] = cm0[j]; m1[j] = cm1[j]; }
        {
            size_t voff = 0;
            #pragma unroll
            for (int kt = 0; kt < 8; ++kt) {
                asm volatile("" : "+s"(voff));
                v16bf A = load_afrag(&h_s[nl][0], kt * 32 + khalf * 8);
                m0 = wmma_bf16(A, load_bfrag(w1h_p, voff, (w    ) * 8 + kt, lane), m0);
                m1 = wmma_bf16(A, load_bfrag(w1h_p, voff, (w + 8) * 8 + kt, lane), m1);
            }
        }
        #pragma unroll
        for (int j = 0; j < 8; ++j) {
            int mr = j + rowOff;
            h1_s[mr][colA] = f2bf(tanh_fast(m0[j]));
            h1_s[mr][colB] = f2bf(tanh_fast(m1[j]));
        }
        __syncthreads();   // h1 ready

        // ---- MLP2: params = h1 @ w2.T + b2 (wave 0, N padded 9->16) ----
        if (w == 0) {
            v8f pa;
            #pragma unroll
            for (int j = 0; j < 8; ++j) pa[j] = pb;
            size_t voff = 0;
            #pragma unroll
            for (int kt = 0; kt < 8; ++kt) {
                asm volatile("" : "+s"(voff));
                v16bf A = load_afrag(&h1_s[nl][0], kt * 32 + khalf * 8);
                pa = wmma_bf16(A, load_bfrag(w2_p, voff, kt, lane), pa);
            }
            #pragma unroll
            for (int j = 0; j < 8; ++j) params_s[j + rowOff][nl] = pa[j];
        }
        __syncthreads();

        // ---- K=3 mixture log-likelihood, masked accumulate ----
        if (tid < 16) {
            const float* p = params_s[tid];
            float target = z_s[tid][t];
            const float Chalf = 0.91893853320467274178f;  // 0.5*log(2*pi)
            float l0 = p[0], l1 = p[1], l2 = p[2];
            float e0 = __expf(-p[6]), e1 = __expf(-p[7]), e2 = __expf(-p[8]);
            float d0 = (target - p[3]) * e0;
            float d1 = (target - p[4]) * e1;
            float d2 = (target - p[5]) * e2;
            float aa0 = -0.5f * d0 * d0 - p[6] - Chalf + l0;
            float aa1 = -0.5f * d1 * d1 - p[7] - Chalf + l1;
            float aa2 = -0.5f * d2 * d2 - p[8] - Chalf + l2;
            float mx  = fmaxf(aa0, fmaxf(aa1, aa2));
            float lse = mx + __logf(__expf(aa0 - mx) + __expf(aa1 - mx) + __expf(aa2 - mx));
            float mq  = fmaxf(l0, fmaxf(l1, l2));
            float lsq = mq + __logf(__expf(l0 - mq) + __expf(l1 - mq) + __expf(l2 - mq));
            if (t < cn) outll += (lse - lsq);
        }
        __syncthreads();
    }

    if (tid < 16) out[base + tid] = outll;
}

// ---------------------------------------------------------------------------
// Launch
// ---------------------------------------------------------------------------
extern "C" void kernel_launch(void* const* d_in, const int* in_sizes, int n_in,
                              void* d_out, int out_size, void* d_ws, size_t ws_size,
                              hipStream_t stream) {
    const float* z    = (const float*)d_in[0];
    const float* c    = (const float*)d_in[1];
    const float* bg   = (const float*)d_in[2];
    const float* m    = (const float*)d_in[3];
    const float* wih  = (const float*)d_in[4];
    const float* whh  = (const float*)d_in[5];
    const float* bih  = (const float*)d_in[6];
    const float* bhh  = (const float*)d_in[7];
    const float* w1   = (const float*)d_in[8];
    const float* b1   = (const float*)d_in[9];
    const float* w2   = (const float*)d_in[10];
    const float* b2   = (const float*)d_in[11];
    float* out = (float*)d_out;

    // Workspace carve-up (all 256B-aligned chunks)
    char* ws = (char*)d_ws;
    float* constih = (float*)ws;                 ws += (size_t)BATCH * G3 * 4;     // 12.6 MB
    float* cbmw1   = (float*)ws;                 ws += (size_t)BATCH * HGRU * 4;   // 4.2 MB
    float* wihz    = (float*)ws;                 ws += 4096;                        // 768*4 padded
    int*   cnt     = (int*)ws;                   ws += (size_t)BATCH * 4;
    unsigned short* whh_p = (unsigned short*)ws; ws += (size_t)WHH_PACK_N * 2;     // 384 KB
    unsigned short* w1h_p = (unsigned short*)ws; ws += (size_t)W1H_PACK_N * 2;     // 128 KB
    unsigned short* w2_p  = (unsigned short*)ws; ws += (size_t)W2_PACK_N * 2;      // 8 KB

    const int packN = WHH_PACK_N + W1H_PACK_N + W2_PACK_N + WIHZ_N;
    flow_pack<<<(packN + 255) / 256, 256, 0, stream>>>(
        whh, w1, w2, wih, whh_p, w1h_p, w2_p, wihz);

    flow_prep<<<BATCH, 256, 0, stream>>>(
        c, bg, m, wih, bih, bhh, w1, b1, constih, cbmw1, cnt);

    flow_main<<<BATCH / 16, 256, 0, stream>>>(
        z, constih, cbmw1, wihz, bhh, cnt, whh_p, w1h_p, w2_p, b2, out);
}
